// build_Generator_54829552501404
// MI455X (gfx1250) — compile-verified, matrix operand order checked by reading, not run on hardware
//
#include <hip/hip_runtime.h>
#include <hip/hip_bf16.h>

// ---------------------------------------------------------------------------
// Generator forward for MI455X (gfx1250, wave32, WMMA, async-to-LDS).
//   h1 = x @ fc_w^T + fc_b              (8192x1024 @ 1024x7056) -> WMMA f16
//   h2 = BNstats(leaky(conv2x2(h1)))    (B,16,14,14), pad bottom/right
//   h3 = leaky(lconv1(BN(h2)))          (B,6,13,13)
//   out = lconv2(h3)                    (B,1,12,12)
// ---------------------------------------------------------------------------

typedef __attribute__((ext_vector_type(16))) _Float16 v16h;
typedef __attribute__((ext_vector_type(8)))  _Float16 v8h;
typedef __attribute__((ext_vector_type(8)))  float    v8f;

#define BATCH   8192
#define LATENT  1024
#define NFC     7056              // 14*14*36
#define H0      14
#define W0      14
#define SLOPE   0.01f
#define ROWH    40                // B-stage LDS row stride in halves (80 B, bank-friendly)

// ---------------------------------------------------------------------------
// K0: f32 -> f16 conversion for GEMM operands
// ---------------------------------------------------------------------------
__global__ void cvt_f16_kernel(const float* __restrict__ x,
                               const float* __restrict__ w,
                               _Float16* __restrict__ xh,
                               _Float16* __restrict__ wh) {
    const int n1 = BATCH * LATENT;
    const int n2 = NFC * LATENT;
    for (int i = blockIdx.x * blockDim.x + threadIdx.x; i < n1 + n2;
         i += gridDim.x * blockDim.x) {
        if (i < n1) xh[i] = (_Float16)x[i];
        else        wh[i - n1] = (_Float16)w[i - n1];
    }
}

// ---------------------------------------------------------------------------
// K1: FC GEMM via v_wmma_f32_16x16x32_f16.
// NT layout: a/b fragments per lane are two 16B chunks at K-offsets khi and
// khi+16 (khi = (lane>>4)*8), rows = lane&15.
// Wave tile: 64(M) x 16(N). B-tile (16x32, shared by all 8 waves of the
// block) is staged into double-buffered LDS by wave 0 with
// global_load_async_to_lds_b128 (ASYNCcnt); A-fragments are register
// double-buffered so WMMAs overlap in-flight loads.
// ---------------------------------------------------------------------------
__device__ __forceinline__ v16h load_frag(const _Float16* __restrict__ p) {
    v8h lo = *(const v8h*)p;          // K = khi .. khi+7
    v8h hi = *(const v8h*)(p + 16);   // K = khi+16 .. khi+23
    v16h r;
#pragma unroll
    for (int i = 0; i < 8; ++i) { r[i] = lo[i]; r[i + 8] = hi[i]; }
    return r;
}

// wave 0: 64 x 16B async transfers of the 16x32 B-tile into LDS (2 per lane)
__device__ __forceinline__ void issue_b_async(_Float16* dst_base,
                                              const _Float16* __restrict__ wh,
                                              int n0, int k0, int lane) {
#pragma unroll
    for (int j = 0; j < 2; ++j) {
        const int t = lane + j * 32;          // 0..63
        const int row = t >> 2, c = t & 3;    // row 0..15, 16B chunk 0..3
        const _Float16* g = wh + (size_t)(n0 + row) * LATENT + k0 + c * 8;
        unsigned lds = (unsigned)(size_t)(dst_base + row * ROWH + c * 8);
        unsigned long long ga = (unsigned long long)(size_t)g;
        asm volatile("global_load_async_to_lds_b128 %0, %1, off"
                     :: "v"(lds), "v"(ga) : "memory");
    }
}

__global__ void fc_wmma_kernel(const _Float16* __restrict__ xh,
                               const _Float16* __restrict__ wh,
                               const float* __restrict__ fc_b,
                               float* __restrict__ h1) {
    __shared__ _Float16 bstage[2][16 * ROWH];   // double-buffered B tile

    const int lane = threadIdx.x & 31;
    const int wave = threadIdx.x >> 5;             // 0..7
    const int n0 = blockIdx.x * 16;                // 441 n-tiles
    const int m0 = blockIdx.y * 512 + wave * 64;   // wave owns 4 m-tiles
    const int r   = lane & 15;
    const int hi  = lane >> 4;
    const int khi = hi * 8;

    const _Float16* ap = xh + (size_t)(m0 + r) * LATENT + khi;

    v8f acc0 = {}, acc1 = {}, acc2 = {}, acc3 = {};

    // prologue: B(k=0) -> LDS buf0, A(k=0) -> registers
    if (wave == 0) issue_b_async(&bstage[0][0], wh, n0, 0, lane);
    v16h ca0 = load_frag(ap);
    v16h ca1 = load_frag(ap + 16 * LATENT);
    v16h ca2 = load_frag(ap + 32 * LATENT);
    v16h ca3 = load_frag(ap + 48 * LATENT);

#pragma unroll 1
    for (int k0 = 0; k0 < LATENT; k0 += 64) {
        // ---------------- phase 1: compute k0, fetch k0+32 ----------------
        if (wave == 0) issue_b_async(&bstage[1][0], wh, n0, k0 + 32, lane);
        v16h na0 = load_frag(ap + k0 + 32);
        v16h na1 = load_frag(ap + 16 * LATENT + k0 + 32);
        v16h na2 = load_frag(ap + 32 * LATENT + k0 + 32);
        v16h na3 = load_frag(ap + 48 * LATENT + k0 + 32);

        if (wave == 0)   // oldest tile (k0) complete; k0+32 may stay in flight
            asm volatile("s_wait_asynccnt 0x2" ::: "memory");
        __syncthreads();
        {
            v16h b = load_frag(&bstage[0][r * ROWH + khi]);
            acc0 = __builtin_amdgcn_wmma_f32_16x16x32_f16(false, ca0, false, b,
                                                          (short)0, acc0, false, false);
            acc1 = __builtin_amdgcn_wmma_f32_16x16x32_f16(false, ca1, false, b,
                                                          (short)0, acc1, false, false);
            acc2 = __builtin_amdgcn_wmma_f32_16x16x32_f16(false, ca2, false, b,
                                                          (short)0, acc2, false, false);
            acc3 = __builtin_amdgcn_wmma_f32_16x16x32_f16(false, ca3, false, b,
                                                          (short)0, acc3, false, false);
        }
        __syncthreads();   // everyone done with buf0 before it is refilled

        // ---------------- phase 2: compute k0+32, fetch k0+64 -------------
        const bool more = (k0 + 64 < LATENT);
        if (more) {
            if (wave == 0) issue_b_async(&bstage[0][0], wh, n0, k0 + 64, lane);
            ca0 = load_frag(ap + k0 + 64);
            ca1 = load_frag(ap + 16 * LATENT + k0 + 64);
            ca2 = load_frag(ap + 32 * LATENT + k0 + 64);
            ca3 = load_frag(ap + 48 * LATENT + k0 + 64);
        }
        if (wave == 0) {
            if (more) asm volatile("s_wait_asynccnt 0x2" ::: "memory");
            else      asm volatile("s_wait_asynccnt 0x0" ::: "memory");
        }
        __syncthreads();
        {
            v16h b = load_frag(&bstage[1][r * ROWH + khi]);
            acc0 = __builtin_amdgcn_wmma_f32_16x16x32_f16(false, na0, false, b,
                                                          (short)0, acc0, false, false);
            acc1 = __builtin_amdgcn_wmma_f32_16x16x32_f16(false, na1, false, b,
                                                          (short)0, acc1, false, false);
            acc2 = __builtin_amdgcn_wmma_f32_16x16x32_f16(false, na2, false, b,
                                                          (short)0, acc2, false, false);
            acc3 = __builtin_amdgcn_wmma_f32_16x16x32_f16(false, na3, false, b,
                                                          (short)0, acc3, false, false);
        }
        __syncthreads();   // everyone done with buf1 before it is refilled
    }

    // C/D layout: element (m,n) lives in vgpr m%8, lane (m/8)*16 + n
    const int n = n0 + r;
    const float bias = fc_b[n];
#pragma unroll
    for (int v = 0; v < 8; ++v) {
        const int ml = v + 8 * hi;
        h1[(size_t)(m0 + ml)      * NFC + n] = acc0[v] + bias;
        h1[(size_t)(m0 + 16 + ml) * NFC + n] = acc1[v] + bias;
        h1[(size_t)(m0 + 32 + ml) * NFC + n] = acc2[v] + bias;
        h1[(size_t)(m0 + 48 + ml) * NFC + n] = acc3[v] + bias;
    }
}

// ---------------------------------------------------------------------------
// K2: conv2x2 (36->16, pad bottom/right) + LeakyReLU + BN partial stats.
// h1 viewed as (B, y, x, c=36); h2 written as (B, 16, 14, 14).
// ---------------------------------------------------------------------------
__global__ void conv_stats_kernel(const float* __restrict__ h1,
                                  const float* __restrict__ cw,
                                  const float* __restrict__ cb,
                                  float* __restrict__ h2,
                                  float* __restrict__ stats) {
    __shared__ float w[16 * 144];   // (oc, c, dy, dx)
    __shared__ float s1[16], s2[16];
    for (int i = threadIdx.x; i < 16 * 144; i += 256) w[i] = cw[i];
    if (threadIdx.x < 16) { s1[threadIdx.x] = 0.f; s2[threadIdx.x] = 0.f; }
    __syncthreads();

    const int idx = blockIdx.x * 256 + threadIdx.x;   // exactly B*196 threads
    const int b = idx / 196, p = idx % 196;
    const int y = p / 14, x = p % 14;

    float acc[16];
#pragma unroll
    for (int oc = 0; oc < 16; ++oc) acc[oc] = cb[oc];

    const float* base = h1 + ((size_t)(b * 14 + y) * 14 + x) * 36;
#pragma unroll
    for (int dy = 0; dy < 2; ++dy) {
        if (y + dy >= H0) break;                 // zero pad (bottom)
#pragma unroll
        for (int dx = 0; dx < 2; ++dx) {
            if (x + dx >= W0) continue;          // zero pad (right)
            const float* pp = base + (dy * 14 + dx) * 36;
            const int kk = dy * 2 + dx;
            for (int c = 0; c < 36; ++c) {
                const float v = pp[c];
#pragma unroll
                for (int oc = 0; oc < 16; ++oc)
                    acc[oc] = fmaf(v, w[oc * 144 + c * 4 + kk], acc[oc]);
            }
        }
    }

#pragma unroll
    for (int oc = 0; oc < 16; ++oc) {
        float v = acc[oc];
        v = (v >= 0.f) ? v : SLOPE * v;
        acc[oc] = v;
        h2[((size_t)b * 16 + oc) * 196 + p] = v;
    }
    // wave32 reduction then LDS atomics, then one global atomic per channel
#pragma unroll
    for (int oc = 0; oc < 16; ++oc) {
        float v = acc[oc], q = v * v;
        for (int off = 16; off > 0; off >>= 1) {
            v += __shfl_xor(v, off, 32);
            q += __shfl_xor(q, off, 32);
        }
        if ((threadIdx.x & 31) == 0) {
            atomicAdd(&s1[oc], v);
            atomicAdd(&s2[oc], q);
        }
    }
    __syncthreads();
    if (threadIdx.x < 16) {
        atomicAdd(&stats[threadIdx.x], s1[threadIdx.x]);
        atomicAdd(&stats[16 + threadIdx.x], s2[threadIdx.x]);
    }
}

// ---------------------------------------------------------------------------
// K3: fold BN (batch stats, biased var) with gamma/beta into scale/shift
// ---------------------------------------------------------------------------
__global__ void bn_finalize_kernel(const float* __restrict__ stats,
                                   const float* __restrict__ gamma,
                                   const float* __restrict__ beta,
                                   float* __restrict__ sc) {
    const int c = threadIdx.x;
    if (c < 16) {
        const float N = (float)BATCH * 196.0f;
        const float mu = stats[c] / N;
        const float var = stats[16 + c] / N - mu * mu;
        const float rs = rsqrtf(var + 1e-5f);
        const float scale = gamma[c] * rs;
        sc[c] = scale;
        sc[16 + c] = beta[c] - mu * scale;
    }
}

// ---------------------------------------------------------------------------
// K4: locally-connected 16->6 (14->13) with BN fused, + LeakyReLU
// lw1: (6,16,13,13,4), k order (0,0),(0,1),(1,0),(1,1)
// ---------------------------------------------------------------------------
__global__ void lconv1_kernel(const float* __restrict__ h2,
                              const float* __restrict__ sc,
                              const float* __restrict__ lw1,
                              float* __restrict__ h3) {
    __shared__ float scale[16], shift[16];
    if (threadIdx.x < 16) {
        scale[threadIdx.x] = sc[threadIdx.x];
        shift[threadIdx.x] = sc[16 + threadIdx.x];
    }
    __syncthreads();

    const int idx = blockIdx.x * 256 + threadIdx.x;   // exactly B*169 threads
    const int b = idx / 169, p = idx % 169;
    const int y = p / 13, x = p % 13;

    float o[6] = {0.f, 0.f, 0.f, 0.f, 0.f, 0.f};
    const float* hb = h2 + (size_t)b * 16 * 196;
    for (int c = 0; c < 16; ++c) {
        const float s = scale[c], t = shift[c];
        const float* pc = hb + c * 196 + y * 14 + x;
        const float v00 = pc[0]  * s + t;
        const float v01 = pc[1]  * s + t;
        const float v10 = pc[14] * s + t;
        const float v11 = pc[15] * s + t;
#pragma unroll
        for (int oo = 0; oo < 6; ++oo) {
            const float* wp = lw1 + ((((size_t)oo * 16 + c) * 13 + y) * 13 + x) * 4;
            float a = fmaf(v00, wp[0], o[oo]);
            a = fmaf(v01, wp[1], a);
            a = fmaf(v10, wp[2], a);
            o[oo] = fmaf(v11, wp[3], a);
        }
    }
#pragma unroll
    for (int oo = 0; oo < 6; ++oo) {
        float v = o[oo];
        v = (v >= 0.f) ? v : SLOPE * v;
        h3[((size_t)b * 6 + oo) * 169 + p] = v;
    }
}

// ---------------------------------------------------------------------------
// K5: locally-connected 6->1 (13->12) -> output (B,1,12,12)
// lw2: (1,6,12,12,4), staged in LDS (13.5 KB)
// ---------------------------------------------------------------------------
__global__ void lconv2_kernel(const float* __restrict__ h3,
                              const float* __restrict__ lw2,
                              float* __restrict__ out) {
    __shared__ float w[6 * 12 * 12 * 4];   // 3456 floats
    for (int i = threadIdx.x; i < 3456; i += 256) w[i] = lw2[i];
    __syncthreads();

    const int idx = blockIdx.x * 256 + threadIdx.x;   // exactly B*144 threads
    const int b = idx / 144, p = idx % 144;
    const int y = p / 12, x = p % 12;

    float acc = 0.f;
#pragma unroll
    for (int c = 0; c < 6; ++c) {
        const float* pc = h3 + ((size_t)b * 6 + c) * 169 + y * 13 + x;
        const float* wp = &w[((c * 12 + y) * 12 + x) * 4];
        acc = fmaf(pc[0],  wp[0], acc);
        acc = fmaf(pc[1],  wp[1], acc);
        acc = fmaf(pc[13], wp[2], acc);
        acc = fmaf(pc[14], wp[3], acc);
    }
    out[idx] = acc;
}

// ---------------------------------------------------------------------------
// launch
// ---------------------------------------------------------------------------
extern "C" void kernel_launch(void* const* d_in, const int* in_sizes, int n_in,
                              void* d_out, int out_size, void* d_ws, size_t ws_size,
                              hipStream_t stream) {
    const float* x      = (const float*)d_in[0];
    const float* fc_w   = (const float*)d_in[1];
    const float* fc_b   = (const float*)d_in[2];
    const float* conv_w = (const float*)d_in[3];
    const float* conv_b = (const float*)d_in[4];
    const float* gamma  = (const float*)d_in[5];
    const float* beta   = (const float*)d_in[6];
    const float* lw1    = (const float*)d_in[7];
    const float* lw2    = (const float*)d_in[8];
    float* out = (float*)d_out;

    // workspace carve-up (all offsets 256B aligned)
    char* ws = (char*)d_ws;
    _Float16* xh  = (_Float16*)(ws);                                    // 16,777,216 B
    _Float16* wh  = (_Float16*)(ws + 16777216);                         // 14,450,688 B
    float* h1     = (float*)(ws + 31227904);                            // 231,211,008 B
    float* h2     = (float*)(ws + 262438912);                           // 102,760,448 B
    float* h3     = (float*)(ws + 365199360);                           //  33,226,752 B
    float* stats  = (float*)(ws + 398426112);                           // 32 sums
    float* scsh   = stats + 32;                                         // 32 scale/shift

    hipMemsetAsync(stats, 0, 64 * sizeof(float), stream);

    // K0: convert operands to f16
    {
        const int n = (BATCH + NFC) * LATENT;   // 15,613,952 -> 60992 blocks exact
        cvt_f16_kernel<<<n / 256, 256, 0, stream>>>(x, fc_w, xh, wh);
    }
    // K1: WMMA GEMM + bias (async-LDS staged B, register double-buffered A)
    {
        dim3 grid(NFC / 16, BATCH / 512);       // (441, 16)
        fc_wmma_kernel<<<grid, 256, 0, stream>>>(xh, wh, fc_b, h1);
    }
    // K2: conv + leaky + BN partial stats
    conv_stats_kernel<<<(BATCH * 196) / 256, 256, 0, stream>>>(h1, conv_w, conv_b,
                                                               h2, stats);
    // K3: BN finalize
    bn_finalize_kernel<<<1, 32, 0, stream>>>(stats, gamma, beta, scsh);
    // K4: BN apply + lconv1 + leaky
    lconv1_kernel<<<(BATCH * 169) / 256, 256, 0, stream>>>(h2, scsh, lw1, h3);
    // K5: lconv2 -> out
    lconv2_kernel<<<(BATCH * 144) / 256, 256, 0, stream>>>(h3, lw2, out);
}
